// BAScheduler_10093173145617
// MI455X (gfx1250) — compile-verified, matrix operand order checked by reading, not run on hardware
//
#include <hip/hip_runtime.h>

typedef __attribute__((ext_vector_type(16))) _Float16 v16h;
typedef __attribute__((ext_vector_type(8)))  float    v8f;

constexpr int GB    = 4096;   // graphs
constexpr int NBAY  = 50;     // bays per graph
constexpr int SDIM  = 16;     // input state size
constexpr int EDIM  = 128;    // embed dim
constexpr int NHEAD = 8;
// D = 16, scale = 1/sqrt(16) = 0.25

// ---------------------------------------------------------------- helpers
__device__ __forceinline__ float eluf(float x) { return x > 0.f ? x : (__expf(x) - 1.f); }
__device__ __forceinline__ float geluf(float x) {
    const float c = 0.7978845608028654f;
    float u = c * (x + 0.044715f * x * x * x);
    return 0.5f * x * (1.f + tanhf(u));
}

// Fused weight value: col c of (W[:,h*16:h*16+16] @ rel[h]) or plain W[k][c]
__device__ __forceinline__ float fuse_val(const float* W, const float* rel,
                                          int k, int col) {
    if (!rel) return W[k * 128 + col];
    int h = col >> 4, e = col & 15;
    float s = 0.f;
    for (int d = 0; d < 16; ++d)
        s += W[k * 128 + h * 16 + d] * rel[h * 256 + d * 16 + e];
    return s;
}

// ------------------------------------------------------------ prep kernels
// Swizzle a [Ksrc x 128] (optionally rel-fused) weight section into WMMA
// B-fragment order: dst[((t*kSteps+s)*32+lane)*16+j]; 8 n-tiles per section.
__global__ __launch_bounds__(256) void prep_swizzle(const float* W, const float* rel,
                                                    _Float16* dst, int Ksrc, int kSteps) {
    int idx = blockIdx.x * blockDim.x + threadIdx.x;
    int per = kSteps * 512;
    if (idx >= 8 * per) return;
    int t = idx / per, rem = idx % per;
    int s = rem >> 9;
    int l = (rem >> 4) & 31;
    int j = rem & 15;
    int col = t * 16 + (l & 15);
    int kb  = s * 32 + ((l & 16) ? 8 : 0);
    int k   = (j < 8) ? (kb + j) : (kb + 16 + (j - 8));
    float v = (k < Ksrc) ? fuse_val(W, rel, k, col) : 0.f;
    dst[idx] = (_Float16)v;
}

// Plain (row-major) fused weight for the block node's VALU dots.
__global__ __launch_bounds__(256) void prep_plain(const float* W, const float* rel,
                                                  _Float16* dst, int Ksrc,
                                                  int dstStride, int colOff) {
    int idx = blockIdx.x * blockDim.x + threadIdx.x;
    if (idx >= Ksrc * 128) return;
    int k = idx >> 7, c = idx & 127;
    dst[k * dstStride + colOff + c] = (_Float16)fuse_val(W, rel, k, c);
}

__global__ void prep_bias(const float* b, const float* rel, float* dst) {
    int c = threadIdx.x; // 128 threads
    float v;
    if (rel) {
        int h = c >> 4, e = c & 15;
        v = 0.f;
        for (int d = 0; d < 16; ++d) v += b[h * 16 + d] * rel[h * 256 + d * 16 + e];
    } else v = b[c];
    dst[c] = v;
}

// ------------------------------------------------------------- WMMA tiles
// Accumulate nSteps K-steps of a 16x16 tile. A in LDS (f16), row stride lda.
__device__ __forceinline__ v8f wtile_h(v8f acc, const _Float16* A, int lda,
                                       const _Float16* Bf, int s0, int nSteps, int lane) {
    int mrow = lane & 15;
    int koff = (lane & 16) ? 8 : 0;
    for (int s = 0; s < nSteps; ++s) {
        v16h a, b;
        const _Float16* ap = A + mrow * lda + s * 32 + koff;
        #pragma unroll
        for (int j = 0; j < 8; ++j) { a[j] = ap[j]; a[j + 8] = ap[16 + j]; }
        const _Float16* bp = Bf + ((s0 + s) * 32 + lane) * 16;
        #pragma unroll
        for (int j = 0; j < 16; ++j) b[j] = bp[j];
        acc = __builtin_amdgcn_wmma_f32_16x16x32_f16(false, a, false, b,
                                                     (short)0, acc, false, false);
    }
    return acc;
}

// Same, but A source is f32 in LDS (converted on fragment build). lda==0 broadcasts row 0.
__device__ __forceinline__ v8f wtile_f(v8f acc, const float* A, int lda,
                                       const _Float16* Bf, int s0, int nSteps, int lane) {
    int mrow = lane & 15;
    int koff = (lane & 16) ? 8 : 0;
    for (int s = 0; s < nSteps; ++s) {
        v16h a, b;
        const float* ap = A + mrow * lda + s * 32 + koff;
        #pragma unroll
        for (int j = 0; j < 8; ++j) { a[j] = (_Float16)ap[j]; a[j + 8] = (_Float16)ap[16 + j]; }
        const _Float16* bp = Bf + ((s0 + s) * 32 + lane) * 16;
        #pragma unroll
        for (int j = 0; j < 16; ++j) b[j] = bp[j];
        acc = __builtin_amdgcn_wmma_f32_16x16x32_f16(false, a, false, b,
                                                     (short)0, acc, false, false);
    }
    return acc;
}

// -------------------------------------------------------------- main kernel
struct KParams {
    const float *x_block, *x_bay, *pairwise;
    const unsigned char* mask;
    const int* action;
    const float *p_bb0, *p_bs0, *p_sb0, *p_bb1, *p_bs1, *p_sb1;
    const float *outWblk0, *outBblk0, *outWblk1, *outBblk1;
    const float *outBbay0, *outBbay1;
    const float *skip_bay, *skip_blk;
    const float *fcW1, *fcB1, *fcB2;
    const float *aB0, *aB1, *aW2, *aB2;
    const float *cW0, *cB0, *cW1, *cB1, *cW2, *cB2;
    const _Float16 *bayFrag0, *bayFrag1, *blkFW0, *blkFW1, *outFrag0, *outFrag1;
    const float *bayBias0, *bayBias1, *blkBias0, *blkBias1;
    const _Float16 *fc2Frag, *a0Frag, *a1Frag;
    float* out;
};

__global__ __launch_bounds__(256) void hgt_graph_kernel(KParams P) {
    extern __shared__ char smem[];
    _Float16* sA  = (_Float16*)(smem);             // [64][128] f16 GEMM A operand
    _Float16* sP  = (_Float16*)(smem + 16384);     // [50][640] f16 bay projections
    float*    sHa = (float*)(smem + 80384);        // [50][128] bay hidden (f32)
    float*    sAg = (float*)(smem + 105984);       // [50][128] aggregation scratch
    _Float16* sH2 = (_Float16*)(smem + 131584);    // [64][128] f16 (h_added / actor h2)
    float*    sM  = (float*)(smem + 147968);       // misc (1024 f32)
    float* hb   = sM;          // [128] block hidden
    float* hbP  = sM + 128;    // [384] block projections (also critic c2)
    float* hbA  = sM + 512;    // [128] block aggregation
    float* pool = sM + 640;    // [128] pooled bays
    float* c1   = sM + 768;    // [128] critic hidden 1
    float* slog = sM + 896;    // [64]  logits

    const int g    = blockIdx.x;
    const int tid  = threadIdx.x;
    const int lane = tid & 31;
    const int wave = tid >> 5;
    const int m0   = (wave >> 1) * 16;   // m-tile rows
    const int ng   = wave & 1;           // n-tile group
    const int moff = (lane & 16) ? 8 : 0;
    // Full-tile fast path: rows m0+moff .. m0+moff+7 all < NBAY
    const bool fullT = (m0 + moff + 7) < NBAY;

    // ---- init: zero-pad A tiles, load layer-0 inputs
    for (int i = tid; i < 64 * 128; i += 256) { sA[i] = (_Float16)0.f; sH2[i] = (_Float16)0.f; }
    for (int i = tid; i < NBAY * SDIM; i += 256) {
        int j = i >> 4, k = i & 15;
        sA[j * 128 + k] = (_Float16)P.x_bay[(g * NBAY + j) * SDIM + k];
    }
    if (tid < SDIM) hb[tid] = P.x_block[g * SDIM + tid];
    __syncthreads();

    const float scale = 0.25f;

    for (int l = 0; l < 2; ++l) {
        const int fin = l ? 128 : 16;
        const int kst = l ? 4 : 1;
        const _Float16* bayFrag = l ? P.bayFrag1 : P.bayFrag0;
        const float*    bayBias = l ? P.bayBias1 : P.bayBias0;
        const _Float16* blkFW   = l ? P.blkFW1   : P.blkFW0;
        const float*    blkBias = l ? P.blkBias1 : P.blkBias0;
        const float* p_bb = l ? P.p_bb1 : P.p_bb0;
        const float* p_bs = l ? P.p_bs1 : P.p_bs0;
        const float* p_sb = l ? P.p_sb1 : P.p_sb0;
        const _Float16* outFrag = l ? P.outFrag1 : P.outFrag0;
        const float* outBbay = l ? P.outBbay1 : P.outBbay0;
        const float* outWblk = l ? P.outWblk1 : P.outWblk0;
        const float* outBblk = l ? P.outBblk1 : P.outBblk0;

        // ---- block projections (Nb=1 -> VALU dots), fused q|k(x)a_sb|v(x)m_sb
        for (int o = tid; o < 384; o += 256) {
            float acc = blkBias[o];
            for (int k = 0; k < fin; ++k) acc += hb[k] * (float)blkFW[k * 384 + o];
            hbP[o] = acc;
        }
        // ---- bay projections: [64,fin] @ [fin,640] fused (q|k*a_bb|v*m_bb|k*a_bs|v*m_bs)
        for (int t = ng; t < 40; t += 2) {
            if (t + 2 < 40) __builtin_prefetch(bayFrag + (t + 2) * kst * 512 + lane * 16, 0, 1);
            v8f acc = {0.f, 0.f, 0.f, 0.f, 0.f, 0.f, 0.f, 0.f};
            acc = wtile_h(acc, sA + m0 * 128, 128, bayFrag + t * kst * 512, 0, kst, lane);
            int n = t * 16 + (lane & 15);
            float bv = bayBias[n];
            if (fullT) {
                #pragma unroll
                for (int r = 0; r < 8; ++r)
                    sP[(m0 + r + moff) * 640 + n] = (_Float16)(acc[r] + bv);
            } else {
                #pragma unroll
                for (int r = 0; r < 8; ++r) {
                    int m = m0 + r + moff;
                    if (m < NBAY) sP[m * 640 + n] = (_Float16)(acc[r] + bv);
                }
            }
        }
        __syncthreads();

        // ---- bay attention: 3 incoming edges (block + up to 2 chain neighbors)
        for (int idx = tid; idx < NBAY * NHEAD; idx += 256) {
            int j = idx >> 3, h = idx & 7;
            const _Float16* q = sP + j * 640 + h * 16;
            float a0 = 0.f, a1 = 0.f, a2 = 0.f;
            for (int e = 0; e < 16; ++e) a0 += (float)q[e] * hbP[128 + h * 16 + e];
            a0 *= p_sb[h] * scale;
            int jm = j > 0 ? j - 1 : 0;
            int jp = j < NBAY - 1 ? j + 1 : NBAY - 1;
            for (int e = 0; e < 16; ++e) {
                a1 += (float)q[e] * (float)sP[jm * 640 + 128 + h * 16 + e];
                a2 += (float)q[e] * (float)sP[jp * 640 + 128 + h * 16 + e];
            }
            a1 *= p_bb[h] * scale; a2 *= p_bb[h] * scale;
            bool has1 = (j > 0), has2 = (j < NBAY - 1);
            float mx = a0;
            if (has1 && a1 > mx) mx = a1;
            if (has2 && a2 > mx) mx = a2;
            float e0 = __expf(a0 - mx);
            float e1 = has1 ? __expf(a1 - mx) : 0.f;
            float e2 = has2 ? __expf(a2 - mx) : 0.f;
            float inv = 1.f / (e0 + e1 + e2 + 1e-16f);
            for (int e = 0; e < 16; ++e) {
                float v = e0 * hbP[256 + h * 16 + e]
                        + e1 * (float)sP[jm * 640 + 256 + h * 16 + e]
                        + e2 * (float)sP[jp * 640 + 256 + h * 16 + e];
                sAg[j * 128 + h * 16 + e] = v * inv;
            }
        }
        // ---- block attention: 50 incoming bs edges (1 head per thread)
        if (tid < NHEAD) {
            int h = tid;
            float ph = p_bs[h] * scale;
            float al[NBAY]; float mx = -1e30f;
            for (int i = 0; i < NBAY; ++i) {
                float d = 0.f;
                for (int e = 0; e < 16; ++e)
                    d += hbP[h * 16 + e] * (float)sP[i * 640 + 384 + h * 16 + e];
                al[i] = d * ph; if (al[i] > mx) mx = al[i];
            }
            float s = 0.f;
            for (int i = 0; i < NBAY; ++i) { al[i] = __expf(al[i] - mx); s += al[i]; }
            float inv = 1.f / (s + 1e-16f);
            for (int e = 0; e < 16; ++e) {
                float acc = 0.f;
                for (int i = 0; i < NBAY; ++i)
                    acc += al[i] * (float)sP[i * 640 + 512 + h * 16 + e];
                hbA[h * 16 + e] = acc * inv;
            }
        }
        __syncthreads();

        // ---- gelu(agg) -> f16 A operand
        for (int i = tid; i < NBAY * 128; i += 256) sA[i] = (_Float16)geluf(sAg[i]);
        if (tid < 128) hbA[tid] = geluf(hbA[tid]);
        __syncthreads();

        // ---- out projections + skip + elu
        float sgb = l ? (1.f / (1.f + __expf(-P.skip_bay[0]))) : 0.f;
        for (int t = ng; t < 8; t += 2) {
            if (t + 2 < 8) __builtin_prefetch(outFrag + (t + 2) * 4 * 512 + lane * 16, 0, 1);
            v8f acc = {0.f, 0.f, 0.f, 0.f, 0.f, 0.f, 0.f, 0.f};
            acc = wtile_h(acc, sA + m0 * 128, 128, outFrag + t * 4 * 512, 0, 4, lane);
            int n = t * 16 + (lane & 15);
            float bv = outBbay[n];
            if (fullT) {
                #pragma unroll
                for (int r = 0; r < 8; ++r) {
                    int m = m0 + r + moff;
                    float o = acc[r] + bv;
                    if (l) o = sgb * o + (1.f - sgb) * sHa[m * 128 + n];
                    sHa[m * 128 + n] = eluf(o);
                }
            } else {
                #pragma unroll
                for (int r = 0; r < 8; ++r) {
                    int m = m0 + r + moff;
                    if (m < NBAY) {
                        float o = acc[r] + bv;
                        if (l) o = sgb * o + (1.f - sgb) * sHa[m * 128 + n];
                        sHa[m * 128 + n] = eluf(o);
                    }
                }
            }
        }
        float sgk = l ? (1.f / (1.f + __expf(-P.skip_blk[0]))) : 0.f;
        for (int o = tid; o < 128; o += 256) {
            float acc = outBblk[o];
            for (int k = 0; k < 128; ++k) acc += hbA[k] * outWblk[k * 128 + o];
            if (l) acc = sgk * acc + (1.f - sgk) * hb[o];
            hb[o] = eluf(acc);
        }
        __syncthreads();
        // rebuild f16 A operand for next layer
        for (int i = tid; i < NBAY * 128; i += 256) sA[i] = (_Float16)sHa[i];
        __syncthreads();
    }

    // ---- pooled bays
    for (int o = tid; o < 128; o += 256) {
        float s = 0.f;
        for (int j = 0; j < NBAY; ++j) s += sHa[j * 128 + o];
        pool[o] = s * (1.f / NBAY);
    }
    // ---- fc layer 1 (K=2, VALU) -> sA
    for (int i = tid; i < NBAY * 128; i += 256) {
        int j = i >> 7, o = i & 127;
        float v = P.pairwise[(g * NBAY + j) * 2 + 0] * P.fcW1[o]
                + P.pairwise[(g * NBAY + j) * 2 + 1] * P.fcW1[128 + o]
                + P.fcB1[o];
        sA[i] = (_Float16)eluf(v);
    }
    __syncthreads();
    // ---- fc layer 2 (WMMA) -> sH2
    for (int t = ng; t < 8; t += 2) {
        v8f acc = {0.f, 0.f, 0.f, 0.f, 0.f, 0.f, 0.f, 0.f};
        acc = wtile_h(acc, sA + m0 * 128, 128, P.fc2Frag + t * 4 * 512, 0, 4, lane);
        int n = t * 16 + (lane & 15);
        float bv = P.fcB2[n];
        if (fullT) {
            #pragma unroll
            for (int r = 0; r < 8; ++r)
                sH2[(m0 + r + moff) * 128 + n] = (_Float16)eluf(acc[r] + bv);
        } else {
            #pragma unroll
            for (int r = 0; r < 8; ++r) {
                int m = m0 + r + moff;
                if (m < NBAY) sH2[m * 128 + n] = (_Float16)eluf(acc[r] + bv);
            }
        }
    }
    __syncthreads();
    // ---- actor layer 0 (K=384 = [h_bays | h_block(bcast) | h_added]) -> sA
    for (int t = ng; t < 8; t += 2) {
        if (t + 2 < 8) __builtin_prefetch(P.a0Frag + (t + 2) * 12 * 512 + lane * 16, 0, 1);
        v8f acc = {0.f, 0.f, 0.f, 0.f, 0.f, 0.f, 0.f, 0.f};
        const _Float16* Fr = P.a0Frag + t * 12 * 512;
        acc = wtile_f(acc, sHa + m0 * 128, 128, Fr, 0, 4, lane);
        acc = wtile_f(acc, hb, 0, Fr, 4, 4, lane);           // broadcast block h
        acc = wtile_h(acc, sH2 + m0 * 128, 128, Fr, 8, 4, lane);
        int n = t * 16 + (lane & 15);
        float bv = P.aB0[n];
        if (fullT) {
            #pragma unroll
            for (int r = 0; r < 8; ++r)
                sA[(m0 + r + moff) * 128 + n] = (_Float16)eluf(acc[r] + bv);
        } else {
            #pragma unroll
            for (int r = 0; r < 8; ++r) {
                int m = m0 + r + moff;
                if (m < NBAY) sA[m * 128 + n] = (_Float16)eluf(acc[r] + bv);
            }
        }
    }
    __syncthreads();
    // ---- actor layer 1 -> sH2
    for (int t = ng; t < 8; t += 2) {
        v8f acc = {0.f, 0.f, 0.f, 0.f, 0.f, 0.f, 0.f, 0.f};
        acc = wtile_h(acc, sA + m0 * 128, 128, P.a1Frag + t * 4 * 512, 0, 4, lane);
        int n = t * 16 + (lane & 15);
        float bv = P.aB1[n];
        if (fullT) {
            #pragma unroll
            for (int r = 0; r < 8; ++r)
                sH2[(m0 + r + moff) * 128 + n] = (_Float16)eluf(acc[r] + bv);
        } else {
            #pragma unroll
            for (int r = 0; r < 8; ++r) {
                int m = m0 + r + moff;
                if (m < NBAY) sH2[m * 128 + n] = (_Float16)eluf(acc[r] + bv);
            }
        }
    }
    __syncthreads();
    // ---- logits + mask
    for (int j = tid; j < NBAY; j += 256) {
        float d = P.aB2[0];
        for (int k = 0; k < 128; ++k) d += (float)sH2[j * 128 + k] * P.aW2[k];
        slog[j] = P.mask[g * NBAY + j] ? d : -1e9f;
    }
    __syncthreads();
    // ---- log-softmax / entropy / logprob
    if (tid == 0) {
        float mx = -1e30f;
        for (int j = 0; j < NBAY; ++j) mx = fmaxf(mx, slog[j]);
        float se = 0.f;
        for (int j = 0; j < NBAY; ++j) se += __expf(slog[j] - mx);
        float lse = mx + logf(se);
        int act = P.action[g];
        P.out[g] = slog[act] - lse;
        float ent = 0.f;
        for (int j = 0; j < NBAY; ++j) {
            if (P.mask[g * NBAY + j]) { float lp = slog[j] - lse; ent -= __expf(lp) * lp; }
        }
        P.out[2 * GB + g] = ent;
    }
    // ---- critic (VALU: [pool | hb] -> 128 -> 128 -> 1)
    for (int o = tid; o < 128; o += 256) {
        float acc = P.cB0[o];
        for (int k = 0; k < 128; ++k) acc += pool[k] * P.cW0[k * 128 + o];
        for (int k = 0; k < 128; ++k) acc += hb[k] * P.cW0[(128 + k) * 128 + o];
        c1[o] = eluf(acc);
    }
    __syncthreads();
    for (int o = tid; o < 128; o += 256) {
        float acc = P.cB1[o];
        for (int k = 0; k < 128; ++k) acc += c1[k] * P.cW1[k * 128 + o];
        hbP[o] = eluf(acc);
    }
    __syncthreads();
    if (tid == 0) {
        float acc = P.cB2[0];
        for (int k = 0; k < 128; ++k) acc += hbP[k] * P.cW2[k];
        P.out[GB + g] = acc;
    }
}

// ---------------------------------------------------------------- launch
extern "C" void kernel_launch(void* const* d_in, const int* in_sizes, int n_in,
                              void* d_out, int out_size, void* d_ws, size_t ws_size,
                              hipStream_t stream) {
    (void)in_sizes; (void)n_in; (void)out_size; (void)ws_size;
    auto F = [&](int i) { return (const float*)d_in[i]; };
    char* ws = (char*)d_ws;

    size_t off = 0;
    auto alloc = [&](size_t bytes) {
        size_t o = off; off = (off + bytes + 255) & ~(size_t)255; return o;
    };
    // B-fragment arrays (f16) + fused biases (f32)
    size_t oBayFrag0 = alloc(5 * 8 * 1 * 512 * 2);
    size_t oBayFrag1 = alloc(5 * 8 * 4 * 512 * 2);
    size_t oBlkFW0   = alloc(16 * 384 * 2);
    size_t oBlkFW1   = alloc(128 * 384 * 2);
    size_t oOutFrag0 = alloc(8 * 4 * 512 * 2);
    size_t oOutFrag1 = alloc(8 * 4 * 512 * 2);
    size_t oFc2      = alloc(8 * 4 * 512 * 2);
    size_t oA0       = alloc(8 * 12 * 512 * 2);
    size_t oA1       = alloc(8 * 4 * 512 * 2);
    size_t oBayB0    = alloc(640 * 4);
    size_t oBayB1    = alloc(640 * 4);
    size_t oBlkB0    = alloc(384 * 4);
    size_t oBlkB1    = alloc(384 * 4);

    auto swz = [&](const float* W, const float* rel, size_t dstOff, int sec,
                   int Ksrc, int kSteps) {
        _Float16* dst = (_Float16*)(ws + dstOff) + (size_t)sec * 8 * kSteps * 512;
        int total = 8 * kSteps * 512;
        prep_swizzle<<<(total + 255) / 256, 256, 0, stream>>>(W, rel, dst, Ksrc, kSteps);
    };
    auto pln = [&](const float* W, const float* rel, size_t dstOff, int colOff, int Ksrc) {
        _Float16* dst = (_Float16*)(ws + dstOff);
        int total = Ksrc * 128;
        prep_plain<<<(total + 255) / 256, 256, 0, stream>>>(W, rel, dst, Ksrc, 384, colOff);
    };
    auto bia = [&](const float* b, const float* rel, size_t dstOff, int colOff) {
        prep_bias<<<1, 128, 0, stream>>>(b, rel, (float*)(ws + dstOff) + colOff);
    };

    // --- param leaf indices (sorted-key pytree order; base 8) ---
    // layer0: proj bay k=28/29 q=30/31 v=32/33; proj blk k=34/35 q=36/37 v=38/39
    //         rel bb a=40 m=41 p=42; bs a=43 m=44 p=45; sb a=46 m=47 p=48
    //         out bay=24/25 blk=26/27; skip bay=49 blk=50
    // layer1: out bay=51/52 blk=53/54; proj bay k=55/56 q=57/58 v=59/60;
    //         blk k=61/62 q=63/64 v=65/66; rel bb=67/68/69 bs=70/71/72 sb=73/74/75
    swz(F(30), nullptr, oBayFrag0, 0, 16, 1);   // q
    swz(F(28), F(40),  oBayFrag0, 1, 16, 1);    // k (x) bb.a
    swz(F(32), F(41),  oBayFrag0, 2, 16, 1);    // v (x) bb.m
    swz(F(28), F(43),  oBayFrag0, 3, 16, 1);    // k (x) bs.a
    swz(F(32), F(44),  oBayFrag0, 4, 16, 1);    // v (x) bs.m
    swz(F(57), nullptr, oBayFrag1, 0, 128, 4);
    swz(F(55), F(67),  oBayFrag1, 1, 128, 4);
    swz(F(59), F(68),  oBayFrag1, 2, 128, 4);
    swz(F(55), F(70),  oBayFrag1, 3, 128, 4);
    swz(F(59), F(71),  oBayFrag1, 4, 128, 4);
    swz(F(24), nullptr, oOutFrag0, 0, 128, 4);
    swz(F(51), nullptr, oOutFrag1, 0, 128, 4);
    swz(F(22), nullptr, oFc2, 0, 128, 4);       // fc W1 (2nd fc layer)
    swz(F(8),  nullptr, oA0, 0, 384, 12);       // actor W0
    swz(F(10), nullptr, oA1, 0, 128, 4);        // actor W1

    pln(F(36), nullptr, oBlkFW0, 0,   16);
    pln(F(34), F(46),  oBlkFW0, 128, 16);
    pln(F(38), F(47),  oBlkFW0, 256, 16);
    pln(F(63), nullptr, oBlkFW1, 0,   128);
    pln(F(61), F(73),  oBlkFW1, 128, 128);
    pln(F(65), F(74),  oBlkFW1, 256, 128);

    bia(F(31), nullptr, oBayB0, 0);   bia(F(29), F(40), oBayB0, 128);
    bia(F(33), F(41),  oBayB0, 256);  bia(F(29), F(43), oBayB0, 384);
    bia(F(33), F(44),  oBayB0, 512);
    bia(F(58), nullptr, oBayB1, 0);   bia(F(56), F(67), oBayB1, 128);
    bia(F(60), F(68),  oBayB1, 256);  bia(F(56), F(70), oBayB1, 384);
    bia(F(60), F(71),  oBayB1, 512);
    bia(F(37), nullptr, oBlkB0, 0);   bia(F(35), F(46), oBlkB0, 128);
    bia(F(39), F(47),  oBlkB0, 256);
    bia(F(64), nullptr, oBlkB1, 0);   bia(F(62), F(73), oBlkB1, 128);
    bia(F(66), F(74),  oBlkB1, 256);

    KParams P;
    P.x_block = F(0); P.x_bay = F(1); P.pairwise = F(2);
    P.mask = (const unsigned char*)d_in[3];
    P.action = (const int*)d_in[4];
    P.p_bb0 = F(42); P.p_bs0 = F(45); P.p_sb0 = F(48);
    P.p_bb1 = F(69); P.p_bs1 = F(72); P.p_sb1 = F(75);
    P.outWblk0 = F(26); P.outBblk0 = F(27);
    P.outWblk1 = F(53); P.outBblk1 = F(54);
    P.outBbay0 = F(25); P.outBbay1 = F(52);
    P.skip_bay = F(76); P.skip_blk = F(77);
    P.fcW1 = F(20); P.fcB1 = F(21); P.fcB2 = F(23);
    P.aB0 = F(9); P.aB1 = F(11); P.aW2 = F(12); P.aB2 = F(13);
    P.cW0 = F(14); P.cB0 = F(15); P.cW1 = F(16); P.cB1 = F(17);
    P.cW2 = F(18); P.cB2 = F(19);
    P.bayFrag0 = (const _Float16*)(ws + oBayFrag0);
    P.bayFrag1 = (const _Float16*)(ws + oBayFrag1);
    P.blkFW0   = (const _Float16*)(ws + oBlkFW0);
    P.blkFW1   = (const _Float16*)(ws + oBlkFW1);
    P.outFrag0 = (const _Float16*)(ws + oOutFrag0);
    P.outFrag1 = (const _Float16*)(ws + oOutFrag1);
    P.bayBias0 = (const float*)(ws + oBayB0);
    P.bayBias1 = (const float*)(ws + oBayB1);
    P.blkBias0 = (const float*)(ws + oBlkB0);
    P.blkBias1 = (const float*)(ws + oBlkB1);
    P.fc2Frag  = (const _Float16*)(ws + oFc2);
    P.a0Frag   = (const _Float16*)(ws + oA0);
    P.a1Frag   = (const _Float16*)(ws + oA1);
    P.out = (float*)d_out;

    const size_t shmem = 152064; // 148.5 KB dynamic LDS (2 WGs per 320KB WGP)
    (void)hipFuncSetAttribute(reinterpret_cast<const void*>(hgt_graph_kernel),
                              hipFuncAttributeMaxDynamicSharedMemorySize, (int)shmem);
    hgt_graph_kernel<<<GB, 256, shmem, stream>>>(P);
}
